// MambaBlock_29300266893967
// MI455X (gfx1250) — compile-verified, matrix operand order checked by reading, not run on hardware
//
#include <hip/hip_runtime.h>
#include <cstdint>

// ---- problem constants (from reference) ----
#define TOK    4096     // B*T = 2*2048
#define NE     2048     // N_EMBD
#define DI     4096     // D_INNER
#define NH     32       // N_HEADS
#define HD     128      // HEAD_DIM
#define DSTATE 128
#define PROJD  8480     // 2*DI + NH + 2*DSTATE
#define NPAD   8576     // PROJD padded to multiple of 128 for WMMA tiling
#define QCH    64       // chunk size
#define NCHUNK 32       // 2048 / 64

typedef __attribute__((ext_vector_type(16))) __bf16 v16bf;
typedef __attribute__((ext_vector_type(8)))  float  v8f;
typedef __attribute__((ext_vector_type(2)))  float  v2f;
typedef unsigned int v4u __attribute__((ext_vector_type(4)));
typedef int          v8i __attribute__((ext_vector_type(8)));
typedef int          v4i __attribute__((ext_vector_type(4)));

__device__ __forceinline__ unsigned short f2bf(float f) {
  unsigned int u = __float_as_uint(f);
  u += 0x7fffu + ((u >> 16) & 1u);          // round-to-nearest-even
  return (unsigned short)(u >> 16);
}
__device__ __forceinline__ float silu_f(float v) { return v / (1.0f + __expf(-v)); }

#define RFL(x) __builtin_amdgcn_readfirstlane((int)(x))

// ---------------------------------------------------------------------------
// TDM: issue a 2-D tile load (bf16 elements) global -> LDS via a D#.
// tile = tile_d1 rows x tile_d0 elems, row stride = stride0 elems.
// Issued once per calling wave; tracked by TENSORcnt.
// ---------------------------------------------------------------------------
__device__ __forceinline__ void tdm_load_tile_2d(
    unsigned lds_off, const void* gptr,
    unsigned tensor_d0, unsigned stride0,
    unsigned tile_d0, unsigned tile_d1)
{
  const unsigned long long ga = (unsigned long long)(uintptr_t)gptr;
  // D# group 0: count=1 | lds_addr | global_addr[56:0] | type=2
  v4u g0 = { (unsigned)RFL(1u),
             (unsigned)RFL(lds_off),
             (unsigned)RFL((unsigned)ga),
             (unsigned)RFL(((unsigned)(ga >> 32) & 0x01FFFFFFu) | (2u << 30)) };
  // D# group 1: data_size=1 (2B); tensor_dim0; tensor_dim1(=1M); tile dims; stride0
  const unsigned td1 = 0x100000u;           // large: tiles are always in-bounds
  v8i g1 = { RFL(1u << 16),
             RFL((tensor_d0 & 0xFFFFu) << 16),
             RFL(((tensor_d0 >> 16) & 0xFFFFu) | ((td1 & 0xFFFFu) << 16)),
             RFL(((td1 >> 16) & 0xFFFFu) | (tile_d0 << 16)),
             RFL(tile_d1),                  // tile_dim1 | tile_dim2(=0)<<16
             RFL(stride0),                  // tensor_dim0_stride[31:0] (elements)
             RFL(0), RFL(0) };
  v4i gz = { 0, 0, 0, 0 };
#if defined(__clang_major__) && (__clang_major__ >= 23)
  v8i gz8 = { 0, 0, 0, 0, 0, 0, 0, 0 };
  __builtin_amdgcn_tensor_load_to_lds(g0, g1, gz, gz, gz8, 0);
#else
  __builtin_amdgcn_tensor_load_to_lds(g0, g1, gz, gz, 0);
#endif
}

// ---------------------------------------------------------------------------
// RMSNorm + cast to bf16
// ---------------------------------------------------------------------------
__global__ __launch_bounds__(256) void rmsnorm_kernel(
    const float* __restrict__ x, const float* __restrict__ scale,
    unsigned short* __restrict__ h)
{
  const int row = blockIdx.x;
  const int tid = threadIdx.x;
  __shared__ float red[256];
  float s = 0.f;
  for (int i = tid; i < NE; i += 256) {
    float v = x[(size_t)row * NE + i];
    s += v * v;
  }
  red[tid] = s;
  __syncthreads();
  for (int off = 128; off > 0; off >>= 1) {
    if (tid < off) red[tid] += red[tid + off];
    __syncthreads();
  }
  const float r = rsqrtf(red[0] / (float)NE + 1e-6f);
  for (int i = tid; i < NE; i += 256)
    h[(size_t)row * NE + i] = f2bf(x[(size_t)row * NE + i] * r * scale[i]);
}

// ---------------------------------------------------------------------------
// Weight prep: W[k][n] (f32) -> Wt[n][k] (bf16), n padded to Npad with zeros
// ---------------------------------------------------------------------------
__global__ void cast_transpose_kernel(const float* __restrict__ W,
                                      unsigned short* __restrict__ Wt,
                                      int K, int N, int Npad)
{
  int k = blockIdx.x * 16 + threadIdx.x;
  int n = blockIdx.y * 16 + threadIdx.y;
  if (k < K && n < Npad)
    Wt[(size_t)n * K + k] = (n < N) ? f2bf(W[(size_t)k * N + n]) : (unsigned short)0;
}

// ---------------------------------------------------------------------------
// bf16 WMMA GEMM:  C[M][N] = A[M][K] * Bt[N][K]^T  (+ optional residual)
// block = 256 thr = 8 waves (4M x 2N), block tile 128x128, wave tile 32x64.
// Tiles staged by the Tensor Data Mover, double-buffered in LDS.
// ---------------------------------------------------------------------------
__global__ __launch_bounds__(256) void gemm_bf16_wmma(
    const unsigned short* __restrict__ A,   // M x K  bf16 row-major
    const unsigned short* __restrict__ Bt,  // N x K  bf16 row-major (B^T)
    float* __restrict__ C,                  // M x N  f32
    const float* __restrict__ Res,          // optional M x N f32 residual
    int M, int N, int K)
{
  __shared__ __align__(16) unsigned short sA[2 * 128 * 32];
  __shared__ __align__(16) unsigned short sB[2 * 128 * 32];

  const int tid  = threadIdx.x;
  const int lane = tid & 31;
  const int wave = tid >> 5;
  const int wm   = wave & 3;          // 0..3  -> M direction (32 rows each)
  const int wn   = wave >> 2;         // 0..1  -> N direction (64 cols each)
  const int m0   = blockIdx.y * 128;
  const int n0   = blockIdx.x * 128;
  const int m16  = lane & 15;
  const int kb   = lane >> 4;         // K sub-block (ISA 16-bit A/B layout)

  v8f acc[2][4] = {};

  // prologue: TDM-stage k0=0 tiles into buffer 0
  if (wave == 0) {
    tdm_load_tile_2d((unsigned)(uintptr_t)&sA[0], A + (size_t)m0 * K,
                     (unsigned)K, (unsigned)K, 32u, 128u);
    tdm_load_tile_2d((unsigned)(uintptr_t)&sB[0], Bt + (size_t)n0 * K,
                     (unsigned)K, (unsigned)K, 32u, 128u);
    __builtin_amdgcn_s_wait_tensorcnt(0);
  }
  __syncthreads();

  int cur = 0;
  for (int k0 = 0; k0 < K; k0 += 32) {
    const int nxt = cur ^ 1;
    const bool more = (k0 + 32 < K);
    if (wave == 0 && more) {          // async prefetch of next K tile via TDM
      tdm_load_tile_2d((unsigned)(uintptr_t)&sA[nxt * 128 * 32],
                       A + (size_t)m0 * K + (k0 + 32),
                       (unsigned)K, (unsigned)K, 32u, 128u);
      tdm_load_tile_2d((unsigned)(uintptr_t)&sB[nxt * 128 * 32],
                       Bt + (size_t)n0 * K + (k0 + 32),
                       (unsigned)K, (unsigned)K, 32u, 128u);
    }

    // fragments per ISA 16-bit layout: lane(m16,kb): elems0..7 = K 8kb..,
    // elems8..15 = K 16+8kb..
    const int cb = cur * 128 * 32;
    union Frag { uint4 u[2]; v16bf v; } fa[2], fb[4];
#pragma unroll
    for (int s = 0; s < 2; ++s) {
      const int r = wm * 32 + s * 16 + m16;
      fa[s].u[0] = *(const uint4*)&sA[cb + r * 32 + kb * 8];
      fa[s].u[1] = *(const uint4*)&sA[cb + r * 32 + 16 + kb * 8];
    }
#pragma unroll
    for (int t = 0; t < 4; ++t) {
      const int r = wn * 64 + t * 16 + m16;
      fb[t].u[0] = *(const uint4*)&sB[cb + r * 32 + kb * 8];
      fb[t].u[1] = *(const uint4*)&sB[cb + r * 32 + 16 + kb * 8];
    }
#pragma unroll
    for (int s = 0; s < 2; ++s)
#pragma unroll
      for (int t = 0; t < 4; ++t)
        acc[s][t] = __builtin_amdgcn_wmma_f32_16x16x32_bf16(
            false, fa[s].v, false, fb[t].v, (short)0, acc[s][t], false, false);

    __syncthreads();                              // all reads of buf[cur] done
    if (wave == 0 && more) __builtin_amdgcn_s_wait_tensorcnt(0);
    __syncthreads();                              // buf[nxt] ready for all
    cur = nxt;
  }

  // epilogue: C/D layout -> VGPR r holds M = r + 8*(lane>=16), N = lane%16
  const int rbase = (lane >> 4) * 8;
  const int cloc  = lane & 15;
#pragma unroll
  for (int s = 0; s < 2; ++s)
#pragma unroll
    for (int t = 0; t < 4; ++t)
#pragma unroll
      for (int r = 0; r < 8; ++r) {
        const int row = m0 + wm * 32 + s * 16 + rbase + r;
        const int col = n0 + wn * 64 + t * 16 + cloc;
        float v = acc[s][t][r];
        if (Res) v += Res[(size_t)row * N + col];
        C[(size_t)row * N + col] = v;
      }
}

// ---------------------------------------------------------------------------
// Depthwise causal conv (D_CONV=4) + bias + SiLU on the ssm_in slice of proj
// ---------------------------------------------------------------------------
__global__ __launch_bounds__(256) void conv_silu_kernel(
    const float* __restrict__ proj, const float* __restrict__ cw,
    const float* __restrict__ cb, float* __restrict__ convx)
{
  const size_t idx = (size_t)blockIdx.x * 256 + threadIdx.x;  // < TOK*DI
  const int cch = (int)(idx & (DI - 1));
  const int rt  = (int)(idx >> 12);
  const int b   = rt >> 11;
  const int t   = rt & 2047;
  float acc = cb[cch];
#pragma unroll
  for (int j = 0; j < 4; ++j) {
    const int ts = t - 3 + j;
    if (ts >= 0)
      acc += cw[j * DI + cch] *
             proj[(size_t)(b * 2048 + ts) * NPAD + DI + cch];
  }
  convx[idx] = silu_f(acc);
}

// ---------------------------------------------------------------------------
// dt = softplus(dt_raw + dt_bias)
// ---------------------------------------------------------------------------
__global__ __launch_bounds__(256) void dt_kernel(
    const float* __restrict__ proj, const float* __restrict__ dt_bias,
    float* __restrict__ dt)
{
  const size_t idx = (size_t)blockIdx.x * 256 + threadIdx.x;  // < TOK*NH
  const int hh  = (int)(idx & (NH - 1));
  const size_t row = idx >> 5;
  const float v = proj[row * NPAD + 2 * DI + hh] + dt_bias[hh];
  dt[idx] = (v > 20.f) ? v : log1pf(__expf(v));
}

// ---------------------------------------------------------------------------
// SSD chunked scan: one workgroup per (batch, head); 128x128 f32 state in LDS,
// sequential over 32 chunks. All four inner products run on the matrix pipe
// via V_WMMA_F32_16X16X4_F32 (A/B = 2 f32/lane, lanes 0-15 K=2kb..2kb+1).
// Dynamic LDS: 115,200 B.
// ---------------------------------------------------------------------------
__global__ __launch_bounds__(256) void ssd_scan_kernel(
    const float* __restrict__ proj, const float* __restrict__ convx,
    const float* __restrict__ dtp, const float* __restrict__ A_log,
    float* __restrict__ yout)
{
  extern __shared__ float smem[];
  float* state = smem;            // 128*128  [n][p]
  float* CB    = state + 16384;   // 64*64, becomes masked W in place
  float* xs    = CB + 4096;       // 64*128   [j][p]
  float* dts   = xs + 8192;       // 64
  float* cum   = dts + 64;        // 64

  const int tid   = threadIdx.x;
  const int lane  = tid & 31;
  const int wv    = tid >> 5;     // 8 waves
  const int m16   = lane & 15;
  const int kb    = lane >> 4;    // 0/1 : K-pair select for f32 WMMA frags
  const int rbase = kb * 8;       // C/D row base per lane half

  const int b = blockIdx.x >> 5;
  const int h = blockIdx.x & 31;
  const float Aval = -__expf(A_log[h]);
  const size_t rowBase = (size_t)b * 2048;
  const int offB = 2 * DI + NH;            // B columns in proj
  const int offC = offB + DSTATE;          // C columns in proj

  for (int i = tid; i < 16384; i += 256) state[i] = 0.0f;
  __syncthreads();

  for (int c = 0; c < NCHUNK; ++c) {
    const int t0 = c * QCH;
    if (tid < QCH) dts[tid] = dtp[(rowBase + t0 + tid) * NH + h];
    for (int idx = tid; idx < QCH * HD; idx += 256) {
      const int j = idx >> 7, p = idx & 127;
      xs[idx] = convx[(rowBase + t0 + j) * DI + h * HD + p];
    }
    __syncthreads();
    if (tid == 0) {                         // inclusive cumsum of A*dt
      float s = 0.f;
      for (int j = 0; j < QCH; ++j) { s += Aval * dts[j]; cum[j] = s; }
    }
    __syncthreads();

    // ---- CB(64x64) = Cc(64x128) @ Bc^T : WMMA f32, 16 tiles, 2 per wave ----
    for (int t = wv; t < 16; t += 8) {
      const int mt = t >> 2, nt = t & 3;
      v8f acc = {};
      const float* Ca = proj + (rowBase + t0 + mt * 16 + m16) * NPAD + offC;
      const float* Bb = proj + (rowBase + t0 + nt * 16 + m16) * NPAD + offB;
      for (int k0 = 0; k0 < DSTATE; k0 += 4) {
        v2f a  = *(const v2f*)&Ca[k0 + 2 * kb];
        v2f bf = *(const v2f*)&Bb[k0 + 2 * kb];
        acc = __builtin_amdgcn_wmma_f32_16x16x4_f32(
            false, a, false, bf, (short)0, acc, false, false);
      }
#pragma unroll
      for (int r = 0; r < 8; ++r)
        CB[(mt * 16 + rbase + r) * 64 + nt * 16 + m16] = acc[r];
    }
    __syncthreads();

    // ---- W[i][j] = (j<=i) ? exp(cum_i-cum_j)*CB*dt_j : 0  (in place) ----
    for (int idx = tid; idx < QCH * QCH; idx += 256) {
      const int i = idx >> 6, j = idx & 63;
      CB[idx] = (j <= i) ? __expf(cum[i] - cum[j]) * CB[idx] * dts[j] : 0.f;
    }
    __syncthreads();

    // ---- Y(64x128) = W @ xs  +  rowscale(exp(cum_i)) * (Cc @ state) ----
    for (int t = wv; t < 32; t += 8) {
      const int mt = t >> 3, nt = t & 7;
      const int pcol = nt * 16 + m16;
      v8f acc = {};
      const float* Wrow = CB + (mt * 16 + m16) * 64;
      for (int k0 = 0; k0 < QCH; k0 += 4) {
        v2f a = *(const v2f*)&Wrow[k0 + 2 * kb];
        v2f bf;
        bf.x = xs[(k0 + 2 * kb) * HD + pcol];
        bf.y = xs[(k0 + 2 * kb + 1) * HD + pcol];
        acc = __builtin_amdgcn_wmma_f32_16x16x4_f32(
            false, a, false, bf, (short)0, acc, false, false);
      }
      v8f acc2 = {};
      const float* Ca = proj + (rowBase + t0 + mt * 16 + m16) * NPAD + offC;
      for (int k0 = 0; k0 < DSTATE; k0 += 4) {
        v2f a = *(const v2f*)&Ca[k0 + 2 * kb];
        v2f bf;
        bf.x = state[(k0 + 2 * kb) * HD + pcol];
        bf.y = state[(k0 + 2 * kb + 1) * HD + pcol];
        acc2 = __builtin_amdgcn_wmma_f32_16x16x4_f32(
            false, a, false, bf, (short)0, acc2, false, false);
      }
#pragma unroll
      for (int r = 0; r < 8; ++r) {
        const int i = mt * 16 + rbase + r;
        yout[(rowBase + t0 + i) * DI + h * HD + pcol] =
            acc[r] + __expf(cum[i]) * acc2[r];
      }
    }
    __syncthreads();                         // state reads done before update

    // ---- state = exp(cum_63)*state + Bc^T @ (w ∘ xs), 64 tiles, 8/wave ----
    {
      const float cd = __expf(cum[QCH - 1]);
      for (int t = wv; t < 64; t += 8) {
        const int mt = t >> 3, nt = t & 7;
        const int n    = mt * 16 + m16;
        const int pcol = nt * 16 + m16;
        v8f acc = {};
        for (int k0 = 0; k0 < QCH; k0 += 4) {
          const int ja = k0 + 2 * kb;
          v2f a;
          a.x = proj[(rowBase + t0 + ja) * NPAD + offB + n];
          a.y = proj[(rowBase + t0 + ja + 1) * NPAD + offB + n];
          const float wa = __expf(cum[QCH - 1] - cum[ja]) * dts[ja];
          const float wb = __expf(cum[QCH - 1] - cum[ja + 1]) * dts[ja + 1];
          v2f bf;
          bf.x = wa * xs[ja * HD + pcol];
          bf.y = wb * xs[(ja + 1) * HD + pcol];
          acc = __builtin_amdgcn_wmma_f32_16x16x4_f32(
              false, a, false, bf, (short)0, acc, false, false);
        }
#pragma unroll
        for (int r = 0; r < 8; ++r) {
          const int idx = (mt * 16 + rbase + r) * HD + pcol;
          state[idx] = cd * state[idx] + acc[r];
        }
      }
    }
    __syncthreads();
  }
}

// ---------------------------------------------------------------------------
// y = (y_ssd + D*convx) * silu(gate), cast bf16 for out_proj GEMM
// ---------------------------------------------------------------------------
__global__ __launch_bounds__(256) void combine_gate_kernel(
    const float* __restrict__ yssd, const float* __restrict__ convx,
    const float* __restrict__ proj, const float* __restrict__ Dp,
    unsigned short* __restrict__ ybf)
{
  const size_t idx = (size_t)blockIdx.x * 256 + threadIdx.x;  // < TOK*DI
  const int ccol = (int)(idx & (DI - 1));
  const size_t row = idx >> 12;
  const float v = yssd[idx] + Dp[ccol] * convx[idx];
  const float g = proj[row * NPAD + ccol];
  ybf[idx] = f2bf(v * silu_f(g));
}

// ---------------------------------------------------------------------------
extern "C" void kernel_launch(void* const* d_in, const int* in_sizes, int n_in,
                              void* d_out, int out_size, void* d_ws, size_t ws_size,
                              hipStream_t stream)
{
  (void)in_sizes; (void)n_in; (void)out_size; (void)ws_size;
  const float* x      = (const float*)d_in[0];
  const float* nscale = (const float*)d_in[1];
  const float* w_in   = (const float*)d_in[2];   // (NE, PROJD)
  const float* conv_w = (const float*)d_in[3];   // (4, DI)
  const float* conv_b = (const float*)d_in[4];
  const float* A_log  = (const float*)d_in[5];
  const float* dt_b   = (const float*)d_in[6];
  const float* Dp     = (const float*)d_in[7];
  const float* w_out  = (const float*)d_in[8];   // (DI, NE)
  float* out = (float*)d_out;

  char* ws = (char*)d_ws;
  size_t off = 0;
  auto carve = [&](size_t bytes) {
    char* p = ws + off;
    off = (off + bytes + 255) & ~(size_t)255;
    return p;
  };
  unsigned short* hbf = (unsigned short*)carve((size_t)TOK * NE * 2);
  unsigned short* Wt1 = (unsigned short*)carve((size_t)NPAD * NE * 2);
  unsigned short* Wt2 = (unsigned short*)carve((size_t)NE * DI * 2);
  float* proj   = (float*)carve((size_t)TOK * NPAD * 4);
  float* convx  = (float*)carve((size_t)TOK * DI * 4);
  float* dtbuf  = (float*)carve((size_t)TOK * NH * 4);
  float* yssd   = (float*)carve((size_t)TOK * DI * 4);
  unsigned short* ybf = (unsigned short*)carve((size_t)TOK * DI * 2);

  // prep: norm+cast, weight cast+transpose (bf16, B^T layout for WMMA)
  rmsnorm_kernel<<<TOK, 256, 0, stream>>>(x, nscale, hbf);
  cast_transpose_kernel<<<dim3(NE / 16, NPAD / 16), dim3(16, 16), 0, stream>>>(
      w_in, Wt1, NE, PROJD, NPAD);
  cast_transpose_kernel<<<dim3(DI / 16, NE / 16), dim3(16, 16), 0, stream>>>(
      w_out, Wt2, DI, NE, NE);

  // GEMM 1: proj = h @ in_proj_w   (M=4096, N=8576(pad), K=2048)
  gemm_bf16_wmma<<<dim3(NPAD / 128, TOK / 128), 256, 0, stream>>>(
      hbf, Wt1, proj, nullptr, TOK, NPAD, NE);

  // conv + SiLU, dt softplus
  conv_silu_kernel<<<(TOK * DI) / 256, 256, 0, stream>>>(proj, conv_w, conv_b, convx);
  dt_kernel<<<(TOK * NH) / 256, 256, 0, stream>>>(proj, dt_b, dtbuf);

  // SSD scan: 64 workgroups (batch x head), WMMA f32 inner products
  ssd_scan_kernel<<<2 * NH, 256, 28800 * sizeof(float), stream>>>(
      proj, convx, dtbuf, A_log, yssd);

  // gating + cast
  combine_gate_kernel<<<(TOK * DI) / 256, 256, 0, stream>>>(yssd, convx, proj, Dp, ybf);

  // GEMM 2: out = residual + y @ out_proj_w  (M=4096, N=2048, K=4096)
  gemm_bf16_wmma<<<dim3(NE / 128, TOK / 128), 256, 0, stream>>>(
      ybf, Wt2, out, x, TOK, NE, DI);
}
